// MulBilstmCRF_33947421508251
// MI455X (gfx1250) — matrix operand order repaired and checked by reading, not compiled, not run on hardware
//
#include <hip/hip_runtime.h>
#include <hip/hip_bf16.h>

typedef __bf16 bf16_t;
typedef __attribute__((ext_vector_type(16))) __bf16 v16bf;
typedef __attribute__((ext_vector_type(8)))  float  v8f;
typedef __attribute__((ext_vector_type(4)))  float  v4f;
typedef int v4i_ __attribute__((vector_size(16)));   // matches async-LDS builtin param

#define B_   256
#define T_   200
#define E_   256
#define H_   256
#define G4_  1024
#define K_   52
#define NROW (T_*B_)      /* 51200 */
#define KSTART 50
#define KSTOP  51

#if defined(__AMDGCN__) && __has_builtin(__builtin_amdgcn_global_load_async_to_lds_b128)
#define HAVE_ASYNC_LDS 1
#else
#define HAVE_ASYNC_LDS 0
#endif

#define AS1 __attribute__((address_space(1)))
#define AS3 __attribute__((address_space(3)))

// ---------------- workspace layout (bytes, all 256-aligned) ----------------
#define OFF_XBF     ((size_t)0)                               // NROW*E bf16
#define OFF_XGF     (OFF_XBF  + (size_t)NROW*E_*2)            // NROW*1024 bf16
#define OFF_XGB     (OFF_XGF  + (size_t)NROW*G4_*2)
#define OFF_WIHF    (OFF_XGB  + (size_t)NROW*G4_*2)
#define OFF_WIHB    (OFF_WIHF + (size_t)G4_*E_*2)
#define OFF_WHHF    (OFF_WIHB + (size_t)G4_*E_*2)
#define OFF_WHHB    (OFF_WHHF + (size_t)G4_*H_*2)
#define OFF_WTAG    (OFF_WHHB + (size_t)G4_*H_*2)             // 64x512 bf16 (rows >=52 zero)
#define OFF_LSTMOUT (OFF_WTAG + (size_t)64*512*2)             // NROW x 512 bf16, row = b*T+t
#define OFF_FEATS   (OFF_LSTMOUT + (size_t)NROW*512*2)        // NROW x 64 f32
#define OFF_BP      (OFF_FEATS + (size_t)NROW*64*4)           // B*T*K u8 backpointers
#define OFF_STATE   (OFF_BP + (((size_t)B_*T_*K_ + 255)/256)*256)
#define OFF_HBF     OFF_STATE                                  // 2 parity * 2 dir * B*H bf16
#define OFF_HF32    (OFF_HBF  + (size_t)2*2*B_*H_*2)           // 2 dir * B*H f32
#define OFF_CF32    (OFF_HF32 + (size_t)2*B_*H_*4)             // 2 dir * B*H f32
#define OFF_SYNC    (OFF_CF32 + (size_t)2*B_*H_*4)             // counters
#define STATE_BYTES ((size_t)(OFF_SYNC + 256 - OFF_STATE))

// ---------------- WMMA helpers (ISA 7.12.2 layouts, wave32) ----------------
__device__ inline v8f wmma_bf16(v16bf a, v16bf b, v8f c) {
  return __builtin_amdgcn_wmma_f32_16x16x32_bf16(false, a, false, b, (short)0, c, false, false);
}

// A fragment: 16x32 bf16 tile of row-major X[M,K] at (m0,k0).
// lane l holds row m0+(l&15); VGPR v pairs: k = {2v | 16+2(v-4)} + 8*(l>>4)
__device__ inline v16bf load_a(const bf16_t* __restrict__ X, int ld, int m0, int k0, int lane) {
  int m  = m0 + (lane & 15);
  int g8 = (lane >> 4) << 3;
  const unsigned* p = (const unsigned*)(X + (size_t)m * ld + k0 + g8);
  union { v16bf v; unsigned u[8]; } ua;
#pragma unroll
  for (int v = 0; v < 8; ++v) ua.u[v] = p[(v < 4) ? v : (v + 4)];
  return ua.v;
}

// B fragment: 32x16 (KxN) where B[k][n] = W[n0+n][k0+k], W row-major [N,K].
// lane l holds column n0+(l&15), k = {j | 16+(j-8)} + 8*(l>>4): two b128 loads.
__device__ inline v16bf load_b(const bf16_t* __restrict__ W, int ld, int n0, int k0, int lane) {
  const bf16_t* p = W + (size_t)(n0 + (lane & 15)) * ld + k0 + ((lane >> 4) << 3);
  union { v16bf v; v4f f[2]; } ub;
  ub.f[0] = *(const v4f*)p;
  ub.f[1] = *(const v4f*)(p + 16);
  return ub.v;
}

// grid-wide barrier (persistent cooperative kernel, 64 co-resident blocks)
__device__ inline void gbar(int* cnt, volatile int* gen, int nblk) {
  __syncthreads();
  if (threadIdx.x == 0) {
    __threadfence();
    int g = *gen;
    if (atomicAdd(cnt, 1) == nblk - 1) {
      *cnt = 0;
      __threadfence();
      atomicAdd((int*)gen, 1);
    } else {
      while (*gen == g) __builtin_amdgcn_s_sleep(2);
    }
  }
  __syncthreads();
}

// ---------------- small prep kernels ----------------
__global__ void k_f32_to_bf16(const float* __restrict__ s, bf16_t* __restrict__ d, int n) {
  int i = blockIdx.x * blockDim.x + threadIdx.x;
  if (i < n) d[i] = (bf16_t)s[i];
}

__global__ void k_wtag_pad(const float* __restrict__ W, bf16_t* __restrict__ d) {
  int i = blockIdx.x * blockDim.x + threadIdx.x;   // 64*512
  if (i >= 64 * 512) return;
  int r = i >> 9, c = i & 511;
  d[i] = (bf16_t)((r < K_) ? W[r * 512 + c] : 0.f);
}

__global__ void k_zero32(unsigned* __restrict__ p, int n) {
  int i = blockIdx.x * blockDim.x + threadIdx.x;
  if (i < n) p[i] = 0u;
}

// emb[t*B+b, e] = char_table[ci[b,t], e] * feat_table[fi[b,t], e]  (bf16)
__global__ void k_embed(const int* __restrict__ ci, const int* __restrict__ fi,
                        const float* __restrict__ ct, const float* __restrict__ ft,
                        bf16_t* __restrict__ x) {
  int n = blockIdx.x;              // t*B + b
  int t = n / B_, b = n % B_;
  int e = threadIdx.x;
  int c = ci[b * T_ + t], f = fi[b * T_ + t];
  x[(size_t)n * E_ + e] = (bf16_t)(ct[(size_t)c * E_ + e] * ft[(size_t)f * E_ + e]);
}

// ------- input projection: xg_d = x @ Wih_d^T + b_d (bf16 out) -------------
// block = (row tile, dir, col-half); 8 waves share the 16x256 A tile in LDS.
__global__ __launch_bounds__(256)
void k_gemm_xg(const bf16_t* __restrict__ x,
               const bf16_t* __restrict__ wf, const bf16_t* __restrict__ wb,
               const float* __restrict__ biasf, const float* __restrict__ biasb,
               bf16_t* __restrict__ xgf, bf16_t* __restrict__ xgb) {
  __shared__ bf16_t xs[16 * E_];                   // 8 KB
  int lane = threadIdx.x & 31;
  int widx = threadIdx.x >> 5;
  int bid  = blockIdx.x;                           // 0..12799
  int rt   = bid >> 2;                             // row tile (16 rows of 51200)
  int sub  = bid & 3;
  int d    = sub >> 1;
  int cg   = (sub & 1) * 8 + widx;                 // 64-column group 0..15
  const bf16_t* W   = d ? wb : wf;
  const float* bias = d ? biasb : biasf;
  bf16_t* out       = d ? xgb : xgf;
  int m0 = rt * 16;

  // stage A tile: 16 rows x 256 cols bf16 = 512 uint4
  {
    const uint4* s4 = (const uint4*)(x + (size_t)m0 * E_);
    uint4* d4 = (uint4*)xs;
    d4[threadIdx.x]       = s4[threadIdx.x];
    d4[threadIdx.x + 256] = s4[threadIdx.x + 256];
  }
  __syncthreads();

  v8f acc[4] = {v8f{}, v8f{}, v8f{}, v8f{}};
#pragma unroll
  for (int k0 = 0; k0 < E_; k0 += 32) {
    v16bf a = load_a(xs, E_, 0, k0, lane);         // ds loads
#pragma unroll
    for (int q = 0; q < 4; ++q) {
      v16bf bb = load_b(W, E_, cg * 64 + q * 16, k0, lane);
      acc[q] = wmma_bf16(a, bb, acc[q]);
    }
  }
  int mrow = m0 + 8 * (lane >> 4);
#pragma unroll
  for (int q = 0; q < 4; ++q) {
    int col  = cg * 64 + q * 16 + (lane & 15);
    float bv = bias[col];
#pragma unroll
    for (int r = 0; r < 8; ++r)
      out[(size_t)(mrow + r) * G4_ + col] = (bf16_t)(acc[q][r] + bv);
  }
}

// ---------------- persistent BiLSTM recurrence ----------------
// 64 blocks * 8 waves = 512 waves; block -> (dir, b-tile, h-half).
// Loop-invariant Whh slice (512 rows x 256 = 256 KB bf16) staged once in the
// 320 KB WGP LDS; per-step 8 KB h tile staged via async-to-LDS when available.
extern __shared__ char lstm_smem[];                // 256 KB Whh + 8 KB h stage

__global__ __launch_bounds__(256)
void k_lstm(const bf16_t* __restrict__ xgf, const bf16_t* __restrict__ xgb,
            const bf16_t* __restrict__ whhf, const bf16_t* __restrict__ whhb,
            const int* __restrict__ len,
            bf16_t* __restrict__ hbf, float* __restrict__ hf32, float* __restrict__ cf32,
            bf16_t* __restrict__ lstm_out, int* __restrict__ sync) {
  bf16_t* whh_s  = (bf16_t*)lstm_smem;             // [4][128][256]
  bf16_t* hstage = (bf16_t*)(lstm_smem + 512 * 256 * 2);  // [16][256]
  int lane = threadIdx.x & 31;
  int w    = blockIdx.x * 8 + (threadIdx.x >> 5);  // 0..511
  int d    = w >> 8;
  int wv   = w & 255;
  int b0   = (wv >> 4) * 16;                       // same for all waves in block
  int hi   = wv & 15;
  int hbase = (blockIdx.x & 1) * 128;              // block's 128 hidden cols
  int hrel0 = (hi & 7) * 16;                       // wave's h tile inside block
  const bf16_t* xg  = d ? xgb : xgf;
  const bf16_t* whh = d ? whhb : whhf;
  int mb = b0 + 8 * (lane >> 4);                   // +r gives batch row
  int nh = hbase + hrel0 + (lane & 15);            // hidden column
  int* cnt = sync;
  volatile int* gen = sync + 1;

  // one-time stage of the block's Whh slice: rows q*256 + hbase + (0..127)
  {
    const uint4* src = (const uint4*)whh;          // 32 uint4 per 256-col row
    uint4* dst = (uint4*)whh_s;
    for (int i = threadIdx.x; i < 512 * 32; i += 256) {
      int row = i >> 5, col = i & 31;
      int q = row >> 7, hrel = row & 127;
      int grow = q * 256 + hbase + hrel;
      dst[i] = src[(size_t)grow * 32 + col];
    }
  }
  __syncthreads();

  for (int step = 0; step < T_; ++step) {
    int t  = d ? (T_ - 1 - step) : step;
    int pc = step & 1;
    const bf16_t* hb_cur = hbf + (size_t)(pc * 2 + d) * B_ * H_;
    bf16_t*       hb_nxt = hbf + (size_t)((pc ^ 1) * 2 + d) * B_ * H_;

    // stage this step's 16 h rows (8 KB, contiguous) into LDS
    {
      const uint4* hsrc = (const uint4*)(hb_cur + (size_t)b0 * H_);
      uint4* hdst = (uint4*)hstage;
#if HAVE_ASYNC_LDS
      {
        int i = threadIdx.x;                       // 256 threads, 512 uint4
        __builtin_amdgcn_global_load_async_to_lds_b128(
            (AS1 v4i_*)(unsigned long long)(hsrc + i),
            (AS3 v4i_*)(unsigned)(unsigned long long)(hdst + i), 0, 0);
        __builtin_amdgcn_global_load_async_to_lds_b128(
            (AS1 v4i_*)(unsigned long long)(hsrc + i + 256),
            (AS3 v4i_*)(unsigned)(unsigned long long)(hdst + i + 256), 0, 0);
      }
      asm volatile("s_wait_asynccnt 0x0" ::: "memory");
#else
      hdst[threadIdx.x]       = hsrc[threadIdx.x];
      hdst[threadIdx.x + 256] = hsrc[threadIdx.x + 256];
#endif
    }
    __syncthreads();

    v8f acc[4] = {v8f{}, v8f{}, v8f{}, v8f{}};
#pragma unroll
    for (int k0 = 0; k0 < H_; k0 += 32) {
      v16bf a = load_a(hstage, H_, 0, k0, lane);            // ds loads
#pragma unroll
      for (int q = 0; q < 4; ++q) {                         // gate q
        v16bf bb = load_b(whh_s, H_, q * 128 + hrel0, k0, lane);  // ds loads
        acc[q] = wmma_bf16(a, bb, acc[q]);
      }
    }
#pragma unroll
    for (int r = 0; r < 8; ++r) {
      int b = mb + r;
      size_t xrow = ((size_t)t * B_ + b) * G4_;
      float gi = acc[0][r] + (float)xg[xrow + 0 * H_ + nh];
      float gf = acc[1][r] + (float)xg[xrow + 1 * H_ + nh];
      float gg = acc[2][r] + (float)xg[xrow + 2 * H_ + nh];
      float go = acc[3][r] + (float)xg[xrow + 3 * H_ + nh];
      gi = 1.f / (1.f + __expf(-gi));
      gf = 1.f / (1.f + __expf(-gf));
      gg = tanhf(gg);
      go = 1.f / (1.f + __expf(-go));
      size_t sidx = (size_t)d * B_ * H_ + (size_t)b * H_ + nh;
      float c_old = cf32[sidx], h_old = hf32[sidx];
      float c_new = gf * c_old + gi * gg;
      float h_new = go * tanhf(c_new);
      float mt    = (t < len[b]) ? 1.f : 0.f;
      float c_sel = mt * c_new + (1.f - mt) * c_old;
      float h_sel = mt * h_new + (1.f - mt) * h_old;
      cf32[sidx] = c_sel;
      hf32[sidx] = h_sel;
      hb_nxt[(size_t)b * H_ + nh] = (bf16_t)h_sel;
      lstm_out[((size_t)b * T_ + t) * 512 + d * H_ + nh] = (bf16_t)(mt * h_new);
    }
    gbar(cnt, gen, gridDim.x);
  }
}

// ---------------- tag projection: feats = lstm_out @ W_tag^T + b_tag -------
__global__ __launch_bounds__(256)
void k_tag(const bf16_t* __restrict__ lo, const bf16_t* __restrict__ wt,
           const float* __restrict__ bt, float* __restrict__ feats) {
  int lane = threadIdx.x & 31;
  int wid  = blockIdx.x * 8 + (threadIdx.x >> 5);  // 0..3199
  int m0   = wid * 16;
  v8f acc[4] = {v8f{}, v8f{}, v8f{}, v8f{}};
#pragma unroll
  for (int k0 = 0; k0 < 512; k0 += 32) {
    if (k0 + 32 < 512) __builtin_prefetch(lo + (size_t)(m0 + (lane & 15)) * 512 + k0 + 32, 0, 0);
    v16bf a = load_a(lo, 512, m0, k0, lane);
#pragma unroll
    for (int q = 0; q < 4; ++q) {
      v16bf bb = load_b(wt, 512, q * 16, k0, lane);
      acc[q] = wmma_bf16(a, bb, acc[q]);
    }
  }
  int mrow = m0 + 8 * (lane >> 4);
#pragma unroll
  for (int q = 0; q < 4; ++q) {
    int col  = q * 16 + (lane & 15);
    float bv = (col < K_) ? bt[col] : 0.f;
#pragma unroll
    for (int r = 0; r < 8; ++r)
      feats[(size_t)(mrow + r) * 64 + col] = acc[q][r] + bv;
  }
}

// ---------------- Viterbi DP + backtrace (one block per batch row) ---------
__global__ __launch_bounds__(64)
void k_viterbi(const float* __restrict__ feats, const float* __restrict__ trans,
               const int* __restrict__ len, unsigned char* __restrict__ bp,
               float* __restrict__ out) {
  __shared__ float ts[K_ * K_];
  __shared__ float part[2][K_];
  int b   = blockIdx.x;
  int tid = threadIdx.x;
  for (int i = tid; i < K_ * K_; i += 64) ts[i] = trans[i];
  __syncthreads();
  int L = len[b];
  if (tid < K_) part[0][tid] = ts[KSTART * K_ + tid] + feats[((size_t)b * T_) * 64 + tid];
  __syncthreads();
  int p = 0;
  for (int t = 1; t < T_; ++t) {
    if (tid < K_) {
      float best = -3.402823e38f;
      int bi = 0;
      for (int i = 0; i < K_; ++i) {
        float v = part[p][i] + ts[i * K_ + tid];
        if (v > best) { best = v; bi = i; }   // first max (matches argmax)
      }
      float nv = best + feats[((size_t)b * T_ + t) * 64 + tid];
      bool mt = (t < L);
      part[p ^ 1][tid] = mt ? nv : part[p][tid];
      bp[((size_t)b * T_ + t) * K_ + tid] = (unsigned char)(mt ? bi : tid);
    }
    __syncthreads();
    p ^= 1;
  }
  if (tid == 0) {
    float best = -3.402823e38f;
    int tag = 0;
    for (int j = 0; j < K_; ++j) {
      float v = part[p][j] + ts[j * K_ + KSTOP];
      if (v > best) { best = v; tag = j; }
    }
    out[b] = best;
    float* dec = out + B_ + (size_t)b * T_;
    int cur = tag;
    dec[T_ - 1] = ((T_ - 1) < L) ? (float)cur : 0.f;
    for (int t = T_ - 1; t >= 1; --t) {
      cur = bp[((size_t)b * T_ + t) * K_ + cur];
      dec[t - 1] = ((t - 1) < L) ? (float)cur : 0.f;
    }
  }
}

// ---------------- launcher ----------------
extern "C" void kernel_launch(void* const* d_in, const int* in_sizes, int n_in,
                              void* d_out, int out_size, void* d_ws, size_t ws_size,
                              hipStream_t stream) {
  (void)in_sizes; (void)n_in; (void)out_size; (void)ws_size;
  const int*   batch_input   = (const int*)d_in[0];
  const int*   batch_feature = (const int*)d_in[1];
  const int*   batch_len     = (const int*)d_in[2];
  const float* char_table    = (const float*)d_in[5];
  const float* feat_table    = (const float*)d_in[6];
  const float* Wih_f         = (const float*)d_in[7];
  const float* Whh_f         = (const float*)d_in[8];
  const float* b_f           = (const float*)d_in[9];
  const float* Wih_b         = (const float*)d_in[10];
  const float* Whh_b         = (const float*)d_in[11];
  const float* b_b           = (const float*)d_in[12];
  const float* W_tag         = (const float*)d_in[13];
  const float* b_tag         = (const float*)d_in[14];
  const float* transitions   = (const float*)d_in[15];
  float* out = (float*)d_out;

  char* ws = (char*)d_ws;
  bf16_t* Xbf   = (bf16_t*)(ws + OFF_XBF);
  bf16_t* XGf   = (bf16_t*)(ws + OFF_XGF);
  bf16_t* XGb   = (bf16_t*)(ws + OFF_XGB);
  bf16_t* WIHf  = (bf16_t*)(ws + OFF_WIHF);
  bf16_t* WIHb  = (bf16_t*)(ws + OFF_WIHB);
  bf16_t* WHHf  = (bf16_t*)(ws + OFF_WHHF);
  bf16_t* WHHb  = (bf16_t*)(ws + OFF_WHHB);
  bf16_t* WTAG  = (bf16_t*)(ws + OFF_WTAG);
  bf16_t* LOUT  = (bf16_t*)(ws + OFF_LSTMOUT);
  float*  FEATS = (float*)(ws + OFF_FEATS);
  unsigned char* BP = (unsigned char*)(ws + OFF_BP);
  bf16_t* HBF   = (bf16_t*)(ws + OFF_HBF);
  float*  HF32  = (float*)(ws + OFF_HF32);
  float*  CF32  = (float*)(ws + OFF_CF32);
  int*    SYNC  = (int*)(ws + OFF_SYNC);

  // weight conversion to bf16
  int nW = G4_ * E_;
  k_f32_to_bf16<<<(nW + 255) / 256, 256, 0, stream>>>(Wih_f, WIHf, nW);
  k_f32_to_bf16<<<(nW + 255) / 256, 256, 0, stream>>>(Wih_b, WIHb, nW);
  k_f32_to_bf16<<<(nW + 255) / 256, 256, 0, stream>>>(Whh_f, WHHf, nW);
  k_f32_to_bf16<<<(nW + 255) / 256, 256, 0, stream>>>(Whh_b, WHHb, nW);
  k_wtag_pad<<<(64 * 512 + 255) / 256, 256, 0, stream>>>(W_tag, WTAG);

  // zero recurrent state + sync counters (ws is poisoned 0xAA)
  int zwords = (int)(STATE_BYTES / 4);
  k_zero32<<<(zwords + 255) / 256, 256, 0, stream>>>((unsigned*)(ws + OFF_STATE), zwords);

  // embedding product -> bf16, time-major (t*B+b, e)
  k_embed<<<NROW, 256, 0, stream>>>(batch_input, batch_feature, char_table, feat_table, Xbf);

  // input projections for both directions (WMMA bf16, A tile in LDS)
  k_gemm_xg<<<12800, 256, 0, stream>>>(Xbf, WIHf, WIHb, b_f, b_b, XGf, XGb);

  // persistent cooperative BiLSTM recurrence:
  // 64 blocks, 256+8 KB dynamic LDS (Whh slice + h stage), grid barrier/step
  size_t lstm_lds = (size_t)512 * 256 * 2 + (size_t)16 * 256 * 2;
  k_lstm<<<64, 256, lstm_lds, stream>>>(XGf, XGb, WHHf, WHHb, batch_len,
                                        HBF, HF32, CF32, LOUT, SYNC);

  // tag projection (WMMA bf16, N padded 52->64)
  k_tag<<<400, 256, 0, stream>>>(LOUT, WTAG, b_tag, FEATS);

  // Viterbi DP + backtrace
  k_viterbi<<<B_, 64, 0, stream>>>(FEATS, transitions, batch_len, BP, out);
}
